// TransformerEncoderLayerSpatial_24446953849400
// MI455X (gfx1250) — matrix operand order repaired and checked by reading, not compiled, not run on hardware
//
#include <hip/hip_runtime.h>
#include <hip/hip_bf16.h>
#include <cstdint>

typedef __attribute__((ext_vector_type(16))) __bf16 v16bf;
typedef __attribute__((ext_vector_type(8)))  __bf16 v8bf;
typedef __attribute__((ext_vector_type(8)))  float  v8f;
typedef int v4i __attribute__((vector_size(4 * sizeof(int))));

static constexpr int BZ = 2, C = 256, Hh = 56, Ww = 56, NQ = Hh * Ww; // 3136
static constexpr int HEADS = 8, HD = 32, FFN = 1024;
static constexpr int ROWS = BZ * NQ;                                  // 6272
static constexpr float SCALING = 0.17677669529663687f;                // 32^-0.5

#if defined(__has_builtin)
#if __has_builtin(__builtin_amdgcn_global_load_async_to_lds_b128) && \
    __has_builtin(__builtin_amdgcn_s_wait_asynccnt)
#define HAVE_ASYNC_LDS 1
#endif
#endif
#ifndef HAVE_ASYNC_LDS
#define HAVE_ASYNC_LDS 0
#endif

// ---------------------------------------------------------------- utilities
__global__ void cvt_f32_bf16(const float* __restrict__ in, __bf16* __restrict__ out, int n) {
  int i = blockIdx.x * 256 + threadIdx.x;
  if (i < n) out[i] = (__bf16)in[i];
}

// transpose src (b,c,h,w)->(b,hw,c) f32 + bf16, and x+posemb variants (bf16)
__global__ void prep_kernel(const float* __restrict__ src,
                            const float* __restrict__ pr,  // (b,56,256) row posemb (over w)
                            const float* __restrict__ pc,  // (b,56,256) col posemb (over h)
                            float* __restrict__ xf, __bf16* __restrict__ xb,
                            __bf16* __restrict__ xprb, __bf16* __restrict__ xpcb) {
  int i = blockIdx.x * 256 + threadIdx.x;            // over BZ*NQ*C
  int cc = i & (C - 1);
  int q  = (i >> 8) % NQ;
  int b  = i / (NQ * C);
  float v = src[(size_t)(b * C + cc) * NQ + q];
  int hh = q / Ww, wj = q % Ww;
  xf[i]   = v;
  xb[i]   = (__bf16)v;
  xprb[i] = (__bf16)(v + pr[(b * Ww + wj) * C + cc]);
  xpcb[i] = (__bf16)(v + pc[(b * Hh + hh) * C + cc]);
}

__device__ __forceinline__ v16bf load_afrag(const __bf16* __restrict__ arow, int kb, int hf) {
  v8bf a0 = *(const v8bf*)(arow + kb + 16 * hf);
  v8bf a1 = *(const v8bf*)(arow + kb + 8 + 16 * hf);
  return __builtin_shufflevector(a0, a1, 0, 1, 2, 3, 4, 5, 6, 7,
                                 8, 9, 10, 11, 12, 13, 14, 15);
}

// ------------------------------------------------- generic WMMA bf16 GEMM
// C[M,N] = act( (A[M,K] * W[N,K]^T + bias[N]) * alpha )
// per-wave 32x32 super-tile = 2x2 WMMA accumulators (A/B fragments reused 2x)
template <bool RELU, bool WF32, bool WBF16>
__global__ void gemm_bf16(const __bf16* __restrict__ A, const __bf16* __restrict__ W,
                          const float* __restrict__ bias, float* __restrict__ Cf,
                          __bf16* __restrict__ Cb, int M, int N, int K, float alpha) {
  int wave = (blockIdx.x * blockDim.x + threadIdx.x) >> 5;
  int lane = threadIdx.x & 31;
  int nsn  = N >> 5;                       // 32-wide super-tiles in N
  int tm = wave / nsn, tn = wave % nsn;
  if (tm * 32 >= M) return;
  int hf = lane >> 4, l15 = lane & 15;

  const __bf16* arow0 = A + (size_t)(tm * 32 + l15) * K;
  const __bf16* arow1 = arow0 + (size_t)16 * K;
  const __bf16* wrow0 = W + (size_t)(tn * 32 + l15) * K;
  const __bf16* wrow1 = wrow0 + (size_t)16 * K;
  v8f acc00 = {}, acc01 = {}, acc10 = {}, acc11 = {};
  for (int kb = 0; kb < K; kb += 32) {
    v16bf af0 = load_afrag(arow0, kb, hf);
    v16bf af1 = load_afrag(arow1, kb, hf);
    v16bf bf0 = *(const v16bf*)(wrow0 + kb + 16 * hf);
    v16bf bf1 = *(const v16bf*)(wrow1 + kb + 16 * hf);
    acc00 = __builtin_amdgcn_wmma_f32_16x16x32_bf16(false, af0, false, bf0,
                                                    (short)0, acc00, false, false);
    acc01 = __builtin_amdgcn_wmma_f32_16x16x32_bf16(false, af0, false, bf1,
                                                    (short)0, acc01, false, false);
    acc10 = __builtin_amdgcn_wmma_f32_16x16x32_bf16(false, af1, false, bf0,
                                                    (short)0, acc10, false, false);
    acc11 = __builtin_amdgcn_wmma_f32_16x16x32_bf16(false, af1, false, bf1,
                                                    (short)0, acc11, false, false);
  }
  int col0 = tn * 32 + l15, col1 = col0 + 16;
  float bv0 = bias[col0], bv1 = bias[col1];
#pragma unroll
  for (int v = 0; v < 8; v++) {
    int row0 = tm * 32 + v + 8 * hf;
    int row1 = row0 + 16;
    float r00 = (acc00[v] + bv0) * alpha;
    float r01 = (acc01[v] + bv1) * alpha;
    float r10 = (acc10[v] + bv0) * alpha;
    float r11 = (acc11[v] + bv1) * alpha;
    if (RELU) {
      r00 = fmaxf(r00, 0.f); r01 = fmaxf(r01, 0.f);
      r10 = fmaxf(r10, 0.f); r11 = fmaxf(r11, 0.f);
    }
    if (WF32) {
      Cf[(size_t)row0 * N + col0] = r00;
      Cf[(size_t)row0 * N + col1] = r01;
      Cf[(size_t)row1 * N + col0] = r10;
      Cf[(size_t)row1 * N + col1] = r11;
    }
    if (WBF16) {
      Cb[(size_t)row0 * N + col0] = (__bf16)r00;
      Cb[(size_t)row0 * N + col1] = (__bf16)r01;
      Cb[(size_t)row1 * N + col0] = (__bf16)r10;
      Cb[(size_t)row1 * N + col1] = (__bf16)r11;
    }
  }
}

// mean-pool projected K over h (over_h=1 -> out (b,w,c)) or over w (-> (b,h,c))
__global__ void pool_mean(const float* __restrict__ full, __bf16* __restrict__ outb, int over_h) {
  int i = blockIdx.x * 256 + threadIdx.x;   // BZ*56*C
  int cc = i & (C - 1);
  int o  = (i >> 8) % 56;
  int b  = i / (56 * C);
  float s = 0.f;
  if (over_h) {
    for (int hh = 0; hh < Hh; hh++) s += full[(size_t)(b * NQ + hh * Ww + o) * C + cc];
  } else {
    for (int wj = 0; wj < Ww; wj++) s += full[(size_t)(b * NQ + o * Ww + wj) * C + cc];
  }
  outb[i] = (__bf16)(s * (1.f / 56.f));
}

// V (b,hw,heads*hd) -> Vt (b,heads,hd,hw)
__global__ void vtrans_kernel(const __bf16* __restrict__ vb, __bf16* __restrict__ vt) {
  int i = blockIdx.x * 256 + threadIdx.x;   // BZ*HEADS*HD*NQ
  int k = i % NQ;
  int d = (i / NQ) % HD;
  int n = (i / (NQ * HD)) % HEADS;
  int b = i / (NQ * HD * HEADS);
  vt[i] = vb[(size_t)(b * NQ + k) * C + n * HD + d];
}

// scores (b,heads,nq,64) = Q(b,nq,h,d) x K(b,56,h,d)^T with masking
__global__ void scores_kernel(const __bf16* __restrict__ Q, const __bf16* __restrict__ Kp,
                              const unsigned char* __restrict__ pm, int pm_stride,
                              float* __restrict__ S) {
  int wave = (blockIdx.x * blockDim.x + threadIdx.x) >> 5;  // 2*8*196*4 waves
  int lane = threadIdx.x & 31;
  int wt = wave & 3;
  int qt = (wave >> 2) % 196;
  int n  = (wave / 784) % HEADS;
  int b  = wave / 6272;
  int hf = lane >> 4, l15 = lane & 15;

  int q    = qt * 16 + l15;
  int col  = wt * 16 + l15;
  int colc = col < 56 ? col : 55;

  const __bf16* arow = Q + ((size_t)(b * NQ + q) * HEADS + n) * HD;
  v16bf af = load_afrag(arow, 0, hf);
  v16bf bf = *(const v16bf*)(Kp + ((size_t)(b * 56 + colc) * HEADS + n) * HD + 16 * hf);
  v8f acc = {};
  acc = __builtin_amdgcn_wmma_f32_16x16x32_bf16(false, af, false, bf,
                                                (short)0, acc, false, false);
  bool masked = (col >= 56) || (pm[b * NQ + col * pm_stride] != 0);
  size_t base = (size_t)(b * HEADS + n) * NQ * 64;
#pragma unroll
  for (int v = 0; v < 8; v++) {
    int row = qt * 16 + v + 8 * hf;
    S[base + (size_t)row * 64 + col] = masked ? -1e30f : acc[v];
  }
}

// one wave per 64-wide row
__global__ void softmax64_kernel(const float* __restrict__ S, __bf16* __restrict__ P) {
  int row  = (blockIdx.x * blockDim.x + threadIdx.x) >> 5;
  int lane = threadIdx.x & 31;
  const float* s = S + (size_t)row * 64;
  float a = s[lane], c = s[lane + 32];
  float m = fmaxf(a, c);
  for (int o = 16; o; o >>= 1) m = fmaxf(m, __shfl_xor(m, o, 32));
  float e0 = __expf(a - m), e1 = __expf(c - m);
  float sum = e0 + e1;
  for (int o = 16; o; o >>= 1) sum += __shfl_xor(sum, o, 32);
  float inv = 1.f / sum;
  __bf16* p = P + (size_t)row * 64;
  p[lane]      = (__bf16)(e0 * inv);
  p[lane + 32] = (__bf16)(e1 * inv);
}

// fused RCDA: out[q,n,d] = sum_{h,w} awc[q,h]*awr[q,w]*V[h,w,n,d]
__global__ void rcda_kernel(const __bf16* __restrict__ Pr, const __bf16* __restrict__ Pc,
                            const __bf16* __restrict__ Vt, __bf16* __restrict__ O) {
  __shared__ __align__(16) __bf16 awr[HEADS][16][64];
  __shared__ __align__(16) __bf16 awc[HEADS][16][64];
  int b = blockIdx.x / 196, qt = blockIdx.x % 196;

#if HAVE_ASYNC_LDS
  // async-DMA the 2x16KB probability tiles into LDS (16B chunks, 8/thread)
  for (int ch = threadIdx.x; ch < 1024; ch += 256) {
    int n = ch >> 7;                // head: 128 x 16B chunks per head
    int r = ch & 127;               // 16B chunk within the head's 2KB block
    size_t goff = ((size_t)(b * HEADS + n) * NQ + qt * 16) * 64 + (size_t)r * 8;
    __builtin_amdgcn_global_load_async_to_lds_b128(
        (__attribute__((address_space(1))) v4i*)(Pr + goff),
        (__attribute__((address_space(3))) v4i*)((char*)&awr[n][0][0] + r * 16), 0, 0);
    __builtin_amdgcn_global_load_async_to_lds_b128(
        (__attribute__((address_space(1))) v4i*)(Pc + goff),
        (__attribute__((address_space(3))) v4i*)((char*)&awc[n][0][0] + r * 16), 0, 0);
  }
  __builtin_amdgcn_s_wait_asynccnt(0);
  __syncthreads();
#else
  for (int i = threadIdx.x; i < HEADS * 16 * 64; i += 256) {
    int c  = i & 63;
    int qq = (i >> 6) & 15;
    int n  = i >> 10;
    size_t off = ((size_t)(b * HEADS + n) * NQ + qt * 16 + qq) * 64 + c;
    awr[n][qq][c] = Pr[off];
    awc[n][qq][c] = Pc[off];
  }
  __syncthreads();
#endif

  int lane = threadIdx.x & 31;
  int n    = threadIdx.x >> 5;           // wave == head
  int hf = lane >> 4, l15 = lane & 15;
  const __bf16* vbase = Vt + (size_t)(b * HEADS + n) * HD * NQ;
  const __bf16* b0p = vbase + (size_t)l15 * NQ;
  const __bf16* b1p = vbase + (size_t)(l15 + 16) * NQ;
  v8f acc0 = {}, acc1 = {};
  for (int kb = 0; kb < NQ; kb += 32) {
    v16bf af;
#pragma unroll
    for (int e = 0; e < 16; e++) {
      int k  = kb + e + 16 * hf;
      int hh = k / Ww;
      int wj = k - hh * Ww;
      af[e] = (__bf16)((float)awc[n][l15][hh] * (float)awr[n][l15][wj]);
    }
    v16bf bf0 = *(const v16bf*)(b0p + kb + 16 * hf);
    v16bf bf1 = *(const v16bf*)(b1p + kb + 16 * hf);
    acc0 = __builtin_amdgcn_wmma_f32_16x16x32_bf16(false, af, false, bf0,
                                                   (short)0, acc0, false, false);
    acc1 = __builtin_amdgcn_wmma_f32_16x16x32_bf16(false, af, false, bf1,
                                                   (short)0, acc1, false, false);
  }
#pragma unroll
  for (int v = 0; v < 8; v++) {
    int q = qt * 16 + v + 8 * hf;
    size_t o = ((size_t)(b * NQ + q) * HEADS + n) * HD;
    O[o + l15]      = (__bf16)acc0[v];
    O[o + 16 + l15] = (__bf16)acc1[v];
  }
}

// residual add + layernorm; optionally writes (b,c,h,w)-transposed f32
template <bool TRANSPOSE_OUT>
__global__ void add_ln_kernel(const float* __restrict__ X, const float* __restrict__ D,
                              const float* __restrict__ g, const float* __restrict__ bta,
                              float* __restrict__ Yf, __bf16* __restrict__ Yb,
                              float* __restrict__ Yt) {
  __shared__ float red[8];
  int row = blockIdx.x, t = threadIdx.x;
  float s = X[(size_t)row * C + t] + D[(size_t)row * C + t];
  int wave = t >> 5, lane = t & 31;

  float v1 = s;
  for (int o = 16; o; o >>= 1) v1 += __shfl_xor(v1, o, 32);
  if (lane == 0) red[wave] = v1;
  __syncthreads();
  float mean = 0.f;
  for (int i = 0; i < 8; i++) mean += red[i];
  mean *= (1.f / 256.f);
  __syncthreads();

  float d = s - mean;
  float v2 = d * d;
  for (int o = 16; o; o >>= 1) v2 += __shfl_xor(v2, o, 32);
  if (lane == 0) red[wave] = v2;
  __syncthreads();
  float var = 0.f;
  for (int i = 0; i < 8; i++) var += red[i];
  var *= (1.f / 256.f);

  float y = d * rsqrtf(var + 1e-5f) * g[t] + bta[t];
  if (Yf) Yf[(size_t)row * C + t] = y;
  if (Yb) Yb[(size_t)row * C + t] = (__bf16)y;
  if (TRANSPOSE_OUT) {
    int b = row / NQ, q = row % NQ;
    Yt[(size_t)(b * C + t) * NQ + q] = y;
  }
}

// ---------------------------------------------------------------- launcher
extern "C" void kernel_launch(void* const* d_in, const int* in_sizes, int n_in,
                              void* d_out, int out_size, void* d_ws, size_t ws_size,
                              hipStream_t stream) {
  const float* src  = (const float*)d_in[0];
  const unsigned char* pmask = (const unsigned char*)d_in[1];
  const float* pr   = (const float*)d_in[2];
  const float* pc   = (const float*)d_in[3];
  const float* wqr  = (const float*)d_in[4];
  const float* bqr  = (const float*)d_in[5];
  const float* wqc  = (const float*)d_in[6];
  const float* bqc  = (const float*)d_in[7];
  const float* wkr  = (const float*)d_in[8];
  const float* bkr  = (const float*)d_in[9];
  const float* wkc  = (const float*)d_in[10];
  const float* bkc  = (const float*)d_in[11];
  const float* wv   = (const float*)d_in[12];
  const float* bv   = (const float*)d_in[13];
  const float* wo   = (const float*)d_in[14];
  const float* bo   = (const float*)d_in[15];
  const float* n1g  = (const float*)d_in[16];
  const float* n1b  = (const float*)d_in[17];
  const float* l1w  = (const float*)d_in[18];
  const float* l1b  = (const float*)d_in[19];
  const float* l2w  = (const float*)d_in[20];
  const float* l2b  = (const float*)d_in[21];
  const float* n2g  = (const float*)d_in[22];
  const float* n2b  = (const float*)d_in[23];

  char* p = (char*)d_ws;
  auto carve = [&](size_t bytes) -> void* {
    void* r = (void*)p;
    p += (bytes + 255) & ~(size_t)255;
    return r;
  };
  const size_t RC = (size_t)ROWS * C;
  __bf16* wqr_b = (__bf16*)carve(65536 * 2);
  __bf16* wqc_b = (__bf16*)carve(65536 * 2);
  __bf16* wkr_b = (__bf16*)carve(65536 * 2);
  __bf16* wkc_b = (__bf16*)carve(65536 * 2);
  __bf16* wv_b  = (__bf16*)carve(65536 * 2);
  __bf16* wo_b  = (__bf16*)carve(65536 * 2);
  __bf16* l1w_b = (__bf16*)carve(262144 * 2);
  __bf16* l2w_b = (__bf16*)carve(262144 * 2);
  float*  x_f   = (float*)carve(RC * 4);
  __bf16* x_b   = (__bf16*)carve(RC * 2);
  __bf16* xpr_b = (__bf16*)carve(RC * 2);
  __bf16* xpc_b = (__bf16*)carve(RC * 2);
  __bf16* qrow_b = (__bf16*)carve(RC * 2);
  __bf16* qcol_b = (__bf16*)carve(RC * 2);
  float*  krf   = (float*)carve(RC * 4);
  float*  kcf   = (float*)carve(RC * 4);
  __bf16* krow_b = (__bf16*)carve((size_t)BZ * 56 * C * 2);
  __bf16* kcol_b = (__bf16*)carve((size_t)BZ * 56 * C * 2);
  __bf16* v_b   = (__bf16*)carve(RC * 2);
  __bf16* vt_b  = (__bf16*)carve(RC * 2);
  const size_t SCN = (size_t)BZ * HEADS * NQ * 64;
  float*  sr    = (float*)carve(SCN * 4);
  float*  sc    = (float*)carve(SCN * 4);
  __bf16* pr_b  = (__bf16*)carve(SCN * 2);
  __bf16* pc_b  = (__bf16*)carve(SCN * 2);
  __bf16* attn_b = (__bf16*)carve(RC * 2);
  float*  attn_f = (float*)carve(RC * 4);
  float*  x1_f  = (float*)carve(RC * 4);
  __bf16* x1_b  = (__bf16*)carve(RC * 2);
  __bf16* f1_b  = (__bf16*)carve((size_t)ROWS * FFN * 2);
  float*  f2_f  = (float*)carve(RC * 4);

  // 1) weight conversions
  cvt_f32_bf16<<<256, 256, 0, stream>>>(wqr, wqr_b, 65536);
  cvt_f32_bf16<<<256, 256, 0, stream>>>(wqc, wqc_b, 65536);
  cvt_f32_bf16<<<256, 256, 0, stream>>>(wkr, wkr_b, 65536);
  cvt_f32_bf16<<<256, 256, 0, stream>>>(wkc, wkc_b, 65536);
  cvt_f32_bf16<<<256, 256, 0, stream>>>(wv, wv_b, 65536);
  cvt_f32_bf16<<<256, 256, 0, stream>>>(wo, wo_b, 65536);
  cvt_f32_bf16<<<1024, 256, 0, stream>>>(l1w, l1w_b, 262144);
  cvt_f32_bf16<<<1024, 256, 0, stream>>>(l2w, l2w_b, 262144);

  // 2) input prep
  prep_kernel<<<6272, 256, 0, stream>>>(src, pr, pc, x_f, x_b, xpr_b, xpc_b);

  // 3) projections: 32x32 super-tiles -> (6272/32)*(N/32) waves, 8 waves/block
  gemm_bf16<false, false, true><<<196, 256, 0, stream>>>(xpr_b, wqr_b, bqr, nullptr, qrow_b,
                                                         ROWS, C, C, SCALING);
  gemm_bf16<false, false, true><<<196, 256, 0, stream>>>(xpc_b, wqc_b, bqc, nullptr, qcol_b,
                                                         ROWS, C, C, SCALING);
  gemm_bf16<false, true, false><<<196, 256, 0, stream>>>(xpr_b, wkr_b, bkr, krf, nullptr,
                                                         ROWS, C, C, 1.f);
  gemm_bf16<false, true, false><<<196, 256, 0, stream>>>(xpc_b, wkc_b, bkc, kcf, nullptr,
                                                         ROWS, C, C, 1.f);
  gemm_bf16<false, false, true><<<196, 256, 0, stream>>>(x_b, wv_b, bv, nullptr, v_b,
                                                         ROWS, C, C, 1.f);

  // 4) k pooling + V transpose
  pool_mean<<<112, 256, 0, stream>>>(krf, krow_b, 1);
  pool_mean<<<112, 256, 0, stream>>>(kcf, kcol_b, 0);
  vtrans_kernel<<<6272, 256, 0, stream>>>(v_b, vt_b);

  // 5) attention scores + softmax (2*8*196*4 waves = 1568 blocks)
  scores_kernel<<<1568, 256, 0, stream>>>(qrow_b, krow_b, pmask, 1, sr);
  scores_kernel<<<1568, 256, 0, stream>>>(qcol_b, kcol_b, pmask, 56, sc);
  softmax64_kernel<<<6272, 256, 0, stream>>>(sr, pr_b);
  softmax64_kernel<<<6272, 256, 0, stream>>>(sc, pc_b);

  // 6) fused RCDA recombination
  rcda_kernel<<<392, 256, 0, stream>>>(pr_b, pc_b, vt_b, attn_b);

  // 7) output projection + LN1
  gemm_bf16<false, true, false><<<196, 256, 0, stream>>>(attn_b, wo_b, bo, attn_f, nullptr,
                                                         ROWS, C, C, 1.f);
  add_ln_kernel<false><<<6272, 256, 0, stream>>>(x_f, attn_f, n1g, n1b, x1_f, x1_b, nullptr);

  // 8) FFN + LN2 (LN2 writes (b,c,h,w) straight to d_out)
  gemm_bf16<true, false, true><<<784, 256, 0, stream>>>(x1_b, l1w_b, l1b, nullptr, f1_b,
                                                        ROWS, FFN, C, 1.f);
  gemm_bf16<false, true, false><<<196, 256, 0, stream>>>(f1_b, l2w_b, l2b, f2_f, nullptr,
                                                         ROWS, C, FFN, 1.f);
  add_ln_kernel<true><<<6272, 256, 0, stream>>>(x1_f, f2_f, n2g, n2b, nullptr, nullptr,
                                                (float*)d_out);
}
